// EncoderRNNImage_35777077576142
// MI455X (gfx1250) — matrix operand order repaired and checked by reading, not compiled, not run on hardware
//
#include <hip/hip_runtime.h>
#include <hip/hip_bf16.h>
#include <cstdint>

// GRU encoder: B=2048, T=26, I=1024, H=512
#define NB 2048
#define NT 26
#define NI 1024
#define NH 512
#define NG 1536  // 3*H

typedef __attribute__((ext_vector_type(16))) _Float16 v16h;
typedef __attribute__((ext_vector_type(8)))  float    v8f;

union FragAB { v16h h; uint4 q[2]; };

__device__ __forceinline__ float sigm(float x) { return 1.0f / (1.0f + __expf(-x)); }

// ---------- fp32 -> fp16 conversion (one-time, bandwidth bound) ----------
__global__ void cvt_f32_f16(const float* __restrict__ in, _Float16* __restrict__ out, int n) {
    int i = blockIdx.x * blockDim.x + threadIdx.x;
    int stride = gridDim.x * blockDim.x;
    for (; i < n; i += stride) out[i] = (_Float16)in[i];
}

// ---------- zero-init h state (f32 master + f16 WMMA copy) ----------
__global__ void init_h(float* __restrict__ h32, _Float16* __restrict__ h16, int n) {
    int i = blockIdx.x * blockDim.x + threadIdx.x;
    if (i < n) { h32[i] = 0.0f; h16[i] = (_Float16)0.0f; }
}

// ---------- Double-buffered WMMA GEMM: C[M,N](f32) = A[M,K](f16) * B[N,K](f16)^T + bias[N]
// 256 threads (8 waves), tile 128x128, K-step 32, 2-deep software pipeline in LDS.
// Wave grid 2x4: wave (wm,wn) owns 64x32 = 4x2 WMMA 16x16 fragments.
__global__ __launch_bounds__(256) void wmma_gemm_nt(
    const _Float16* __restrict__ A, const _Float16* __restrict__ Bm,
    const float* __restrict__ bias, float* __restrict__ C,
    int M, int N, int K)
{
    // K-major tiles, row stride 40 halves (80B, 16B aligned), double-buffered
    __shared__ _Float16 As[2][128 * 40];
    __shared__ _Float16 Bs[2][128 * 40];

    const int tid  = threadIdx.x;
    const int lane = tid & 31;
    const int wave = tid >> 5;
    const int wm   = wave >> 2;   // 0..1  (64 rows each)
    const int wn   = wave & 3;    // 0..3  (32 cols each)
    const int m0   = blockIdx.y * 128;
    const int n0   = blockIdx.x * 128;

    v8f acc[4][2];
#pragma unroll
    for (int i = 0; i < 4; ++i)
#pragma unroll
        for (int j = 0; j < 2; ++j)
#pragma unroll
            for (int r = 0; r < 8; ++r) acc[i][j][r] = 0.0f;

    // Global->LDS staging: 128 rows x 32 halves = 512 16B-chunks, 2 per thread.
    const int r0 = tid >> 2,          kc0 = (tid & 3) * 8;
    const int r1 = (tid + 256) >> 2,  kc1 = (tid & 3) * 8;  // (tid+256)&3 == tid&3

    // ISA fragment addressing (CDNA5 ISA 7.12.2, wave32):
    const int lrow  = lane & 15;
    const int lhalf = lane >> 4;
    const int akb   = lhalf * 8;      // A kbase: 0 or 8
    const int bkb   = lhalf * 8;      // B kbase: 0 or 8

    uint4 ra0, ra1, rb0, rb1;
    auto gload = [&](int k0) {
        ra0 = *(const uint4*)(A  + (size_t)(m0 + r0) * K + k0 + kc0);
        ra1 = *(const uint4*)(A  + (size_t)(m0 + r1) * K + k0 + kc1);
        rb0 = *(const uint4*)(Bm + (size_t)(n0 + r0) * K + k0 + kc0);
        rb1 = *(const uint4*)(Bm + (size_t)(n0 + r1) * K + k0 + kc1);
    };
    auto sstore = [&](int p) {
        *(uint4*)(&As[p][r0 * 40 + kc0]) = ra0;
        *(uint4*)(&As[p][r1 * 40 + kc1]) = ra1;
        *(uint4*)(&Bs[p][r0 * 40 + kc0]) = rb0;
        *(uint4*)(&Bs[p][r1 * 40 + kc1]) = rb1;
    };

    // prologue: tile 0 -> buf 0; start loading tile 1
    gload(0);
    sstore(0);
    if (K > 32) gload(32);
    __syncthreads();

    int p = 0;
    for (int k0 = 0; k0 < K; k0 += 32) {
        // compute from buffer p
        FragAB bf[2];
#pragma unroll
        for (int j = 0; j < 2; ++j) {
            const int nr = wn * 32 + j * 16 + lrow;
            bf[j].q[0] = *(const uint4*)(&Bs[p][nr * 40 + bkb]);
            bf[j].q[1] = *(const uint4*)(&Bs[p][nr * 40 + bkb + 16]);
        }
#pragma unroll
        for (int i = 0; i < 4; ++i) {
            FragAB af;
            const int mr = wm * 64 + i * 16 + lrow;
            af.q[0] = *(const uint4*)(&As[p][mr * 40 + akb]);
            af.q[1] = *(const uint4*)(&As[p][mr * 40 + 16 + akb]);
#pragma unroll
            for (int j = 0; j < 2; ++j) {
                acc[i][j] = __builtin_amdgcn_wmma_f32_16x16x32_f16(
                    false, af.h, false, bf[j].h, (short)0, acc[i][j], false, false);
            }
        }
        // pipeline: park tile k0+32 (already in regs) into other buffer, start k0+64
        const int kn = k0 + 32;
        if (kn < K) {
            sstore(p ^ 1);
            if (kn + 32 < K) gload(kn + 32);
            __syncthreads();
            p ^= 1;
        }
    }

    // Epilogue: C/D layout — lanes 0-15: M=r, N=lane; lanes 16-31: M=8+r, N=lane-16
#pragma unroll
    for (int i = 0; i < 4; ++i) {
        const int mbase = m0 + wm * 64 + i * 16 + lhalf * 8;
#pragma unroll
        for (int j = 0; j < 2; ++j) {
            const int n  = n0 + wn * 32 + j * 16 + lrow;
            const float bv = bias[n];
#pragma unroll
            for (int r = 0; r < 8; ++r) {
                C[(size_t)(mbase + r) * N + n] = acc[i][j][r] + bv;
            }
        }
    }
}

// ---------- Fused recurrent step: hg = h @ w_hh^T, gates, h-update, output write.
// Block: 256 threads (8 waves), tile 64(batch) x 64(hidden), all 3 gates per tile.
// Wave (wm 0..3: 16 rows, wn 0..1: 32 hidden cols): 1 A-frag x (3 gates x 2) B-frags = 6 WMMA/K-step.
// acc[r], acc[z], acc[n] land in the same lanes at matching (m,h) -> lane-local gate fusion.
// Reads H16in/H32in, writes H16out/H32out (ping-pong between steps to avoid cross-block RAW).
__global__ __launch_bounds__(256) void gru_step_fused(
    const _Float16* __restrict__ H16in, const _Float16* __restrict__ Whh,
    const float* __restrict__ bhh, const float* __restrict__ Xg,
    const float* __restrict__ H32in, float* __restrict__ H32out,
    _Float16* __restrict__ H16out, float* __restrict__ out, int t)
{
    // 4 tiles (h, w_r, w_z, w_n), each 64 rows x 40-half stride
    __shared__ _Float16 S[4 * 64 * 40];

    const int tid  = threadIdx.x;
    const int lane = tid & 31;
    const int wave = tid >> 5;
    const int wm   = wave >> 1;   // 0..3 (16 batch rows each)
    const int wn   = wave & 1;    // 0..1 (32 hidden cols each)
    const int m0   = blockIdx.y * 64;   // batch tile
    const int n0   = blockIdx.x * 64;   // hidden tile

    const int rowc = tid >> 2;          // 0..63: one chunk per thread per tile
    const int kcc  = (tid & 3) * 8;

    const int lrow  = lane & 15;
    const int lhalf = lane >> 4;
    const int akb   = lhalf * 8;
    const int bkb   = lhalf * 8;

    v8f acc[3][2];
#pragma unroll
    for (int g = 0; g < 3; ++g)
#pragma unroll
        for (int j = 0; j < 2; ++j)
#pragma unroll
            for (int r = 0; r < 8; ++r) acc[g][j][r] = 0.0f;

    for (int k0 = 0; k0 < NH; k0 += 32) {
        uint4 l0 = *(const uint4*)(H16in + (size_t)(m0 + rowc) * NH + k0 + kcc);
        uint4 l1 = *(const uint4*)(Whh + (size_t)(0 * NH + n0 + rowc) * NH + k0 + kcc);
        uint4 l2 = *(const uint4*)(Whh + (size_t)(1 * NH + n0 + rowc) * NH + k0 + kcc);
        uint4 l3 = *(const uint4*)(Whh + (size_t)(2 * NH + n0 + rowc) * NH + k0 + kcc);
        __syncthreads();
        *(uint4*)(&S[0 * 2560 + rowc * 40 + kcc]) = l0;
        *(uint4*)(&S[1 * 2560 + rowc * 40 + kcc]) = l1;
        *(uint4*)(&S[2 * 2560 + rowc * 40 + kcc]) = l2;
        *(uint4*)(&S[3 * 2560 + rowc * 40 + kcc]) = l3;
        __syncthreads();

        FragAB af;
        const int mr = wm * 16 + lrow;
        af.q[0] = *(const uint4*)(&S[mr * 40 + akb]);
        af.q[1] = *(const uint4*)(&S[mr * 40 + 16 + akb]);
#pragma unroll
        for (int g = 0; g < 3; ++g) {
#pragma unroll
            for (int j = 0; j < 2; ++j) {
                FragAB bf;
                const int base = (g + 1) * 2560 + (wn * 32 + j * 16 + lrow) * 40;
                bf.q[0] = *(const uint4*)(&S[base + bkb]);
                bf.q[1] = *(const uint4*)(&S[base + bkb + 16]);
                acc[g][j] = __builtin_amdgcn_wmma_f32_16x16x32_f16(
                    false, af.h, false, bf.h, (short)0, acc[g][j], false, false);
            }
        }
    }

    // Lane-local fused GRU gates + state update
    const int mbase = m0 + wm * 16 + lhalf * 8;
#pragma unroll
    for (int j = 0; j < 2; ++j) {
        const int hc = n0 + wn * 32 + j * 16 + lrow;   // hidden col in [0,512)
        const float br = bhh[hc];
        const float bz = bhh[NH + hc];
        const float bn = bhh[2 * NH + hc];
#pragma unroll
        for (int r = 0; r < 8; ++r) {
            const int m = mbase + r;
            const size_t xrow = ((size_t)m * NT + t) * NG;
            const float xr = Xg[xrow + hc];
            const float xz = Xg[xrow + NH + hc];
            const float xn = Xg[xrow + 2 * NH + hc];
            const float rg = sigm(xr + acc[0][j][r] + br);
            const float zg = sigm(xz + acc[1][j][r] + bz);
            const float nn = tanhf(xn + rg * (acc[2][j][r] + bn));
            const float hp = H32in[(size_t)m * NH + hc];
            const float hv = (1.0f - zg) * nn + zg * hp;
            H32out[(size_t)m * NH + hc] = hv;
            H16out[(size_t)m * NH + hc] = (_Float16)hv;
            out[((size_t)m * NT + t) * NH + hc] = hv;               // enc_outputs[m,t,:]
            if (t == NT - 1) out[(size_t)NB * NT * NH + (size_t)m * NH + hc] = hv;  // hidden
        }
    }
}

extern "C" void kernel_launch(void* const* d_in, const int* in_sizes, int n_in,
                              void* d_out, int out_size, void* d_ws, size_t ws_size,
                              hipStream_t stream) {
    const float* img  = (const float*)d_in[0];   // [B,T,I]
    const float* w_ih = (const float*)d_in[1];   // [3H,I]
    const float* w_hh = (const float*)d_in[2];   // [3H,H]
    const float* b_ih = (const float*)d_in[3];   // [3H]
    const float* b_hh = (const float*)d_in[4];   // [3H]
    float* out = (float*)d_out;                  // enc_outputs [B,T,H] ++ hidden [1,B,H]

    char* ws = (char*)d_ws;
    size_t off = 0;
    auto carve = [&](size_t bytes) {
        void* p = ws + off;
        off = (off + bytes + 255) & ~(size_t)255;
        return p;
    };
    _Float16* A16   = (_Float16*)carve((size_t)NB * NT * NI * 2);  // img as f16 (L2-resident)
    _Float16* Wih16 = (_Float16*)carve((size_t)NG * NI * 2);
    _Float16* Whh16 = (_Float16*)carve((size_t)NG * NH * 2);
    float*    Xg    = (float*)   carve((size_t)NB * NT * NG * 4);  // x-gates, f32
    float*    H32a  = (float*)   carve((size_t)NB * NH * 4);       // ping-pong f32 h
    float*    H32b  = (float*)   carve((size_t)NB * NH * 4);
    _Float16* H16a  = (_Float16*)carve((size_t)NB * NH * 2);       // ping-pong f16 h
    _Float16* H16b  = (_Float16*)carve((size_t)NB * NH * 2);

    // 1) precision conversion
    cvt_f32_f16<<<2048, 256, 0, stream>>>(img,  A16,   NB * NT * NI);
    cvt_f32_f16<<<512,  256, 0, stream>>>(w_ih, Wih16, NG * NI);
    cvt_f32_f16<<<512,  256, 0, stream>>>(w_hh, Whh16, NG * NH);
    init_h<<<(NB * NH + 255) / 256, 256, 0, stream>>>(H32a, H16a, NB * NH);

    // 2) input projection: Xg = img @ w_ih^T + b_ih   (M=53248, N=1536, K=1024)
    {
        dim3 grid(NG / 128, (NB * NT) / 128);  // 12 x 416
        wmma_gemm_nt<<<grid, 256, 0, stream>>>(A16, Wih16, b_ih, Xg, NB * NT, NG, NI);
    }

    // 3) fused serial recurrence: one launch per step, ping-pong h buffers
    {
        dim3 grid(NH / 64, NB / 64);  // 8 x 32 = 256 blocks
        for (int t = 0; t < NT; ++t) {
            const bool even = ((t & 1) == 0);
            gru_step_fused<<<grid, 256, 0, stream>>>(
                even ? H16a : H16b, Whh16, b_hh, Xg,
                even ? H32a : H32b, even ? H32b : H32a,
                even ? H16b : H16a, out, t);
        }
    }
}
